// GatedMoEFFN_49160195670636
// MI455X (gfx1250) — compile-verified, matrix operand order checked by reading, not run on hardware
//
#include <hip/hip_runtime.h>
#include <hip/hip_bf16.h>

#define DM   1024   // d_model
#define DF   4096   // d_ff
#define NE   8      // experts
#define NTOK 2048   // tokens (2*1024)
#define CAP  2048   // per-expert list capacity
#define RS   40     // LDS row stride in ushorts (80B: 16B aligned, 20-bank skew)

typedef __attribute__((ext_vector_type(16))) __bf16 v16bf;
typedef __attribute__((ext_vector_type(2)))  __bf16 v2bf;
typedef __attribute__((ext_vector_type(8)))  float  v8f;

union FragU {
    v16bf v;
    uint4 q[2];
};

// Native fp32 -> bf16 (RNE) conversions: let the compiler pick v_cvt_pk_bf16_f32.
__device__ __forceinline__ unsigned short f2bf(float f) {
    return __builtin_bit_cast(unsigned short, (__bf16)f);
}

__device__ __forceinline__ unsigned pack_bf2(float lo, float hi) {
    v2bf v = {(__bf16)lo, (__bf16)hi};
    return __builtin_bit_cast(unsigned, v);
}

__device__ __forceinline__ float gelu_exact(float v) {
    return 0.5f * v * (1.0f + erff(v * 0.70710678118654752f));
}

// Load one 16x32 bf16 fragment (A-layout; B uses identical layout on the
// N-major transposed LDS tile). Two conflict-free ds_load_b128 per lane.
__device__ __forceinline__ v16bf ld_frag(const unsigned short* lds, int row0, int lane) {
    FragU f;
    const unsigned short* p = lds + (size_t)(row0 + (lane & 15)) * RS + ((lane >> 4) * 8);
    f.q[0] = *(const uint4*)(p);
    f.q[1] = *(const uint4*)(p + 16);
    return f.v;
}

// Async DMA: 16 bytes global -> LDS per lane (ASYNCcnt tracked).
__device__ __forceinline__ void async_copy_b128(unsigned lds_off, const void* gptr) {
    asm volatile("global_load_async_to_lds_b128 %0, %1, off"
                 :: "v"(lds_off), "v"(gptr) : "memory");
}

__device__ __forceinline__ void wait_async0() {
    asm volatile("s_wait_asynccnt 0x0" ::: "memory");
}

// ---------------------------------------------------------------- zero init
__global__ __launch_bounds__(256) void moe_zero_kernel(float* __restrict__ out,
                                                       int n, int* __restrict__ counts) {
    int i = blockIdx.x * blockDim.x + threadIdx.x;
    for (; i < n; i += gridDim.x * blockDim.x) out[i] = 0.0f;
    if (blockIdx.x == 0 && threadIdx.x < NE) counts[threadIdx.x] = 0;
}

// ---------------------------------------------------------------- router
// One wave per token: 8 logits, cross-lane reduce, top-2 + softmax, append.
__global__ __launch_bounds__(256) void moe_router_kernel(
    const float* __restrict__ x, const float* __restrict__ Wg,
    const float* __restrict__ bg, int* __restrict__ counts,
    int* __restrict__ list_slot, float* __restrict__ list_w) {
    const int token = blockIdx.x * 8 + (threadIdx.x >> 5);
    const int lane  = threadIdx.x & 31;
    if (token >= NTOK) return;

    float acc[NE];
#pragma unroll
    for (int e = 0; e < NE; ++e) acc[e] = 0.0f;

    const float* xr = x + (size_t)token * DM;
    for (int d = lane; d < DM; d += 32) {
        float xv = xr[d];
        const float4* wr = (const float4*)(Wg + (size_t)d * NE);
        float4 w0 = wr[0], w1 = wr[1];
        acc[0] += xv * w0.x; acc[1] += xv * w0.y;
        acc[2] += xv * w0.z; acc[3] += xv * w0.w;
        acc[4] += xv * w1.x; acc[5] += xv * w1.y;
        acc[6] += xv * w1.z; acc[7] += xv * w1.w;
    }
#pragma unroll
    for (int off = 16; off > 0; off >>= 1) {
#pragma unroll
        for (int e = 0; e < NE; ++e) acc[e] += __shfl_down(acc[e], off, 32);
    }
    if (lane == 0) {
        float l[NE];
#pragma unroll
        for (int e = 0; e < NE; ++e) l[e] = acc[e] + bg[e];
        int i0 = 0;
#pragma unroll
        for (int e = 1; e < NE; ++e) if (l[e] > l[i0]) i0 = e;
        int i1 = (i0 == 0) ? 1 : 0;
#pragma unroll
        for (int e = 0; e < NE; ++e) if (e != i0 && l[e] > l[i1]) i1 = e;

        float ex = __expf(l[i1] - l[i0]);      // softmax over the top-2
        float w0 = 1.0f / (1.0f + ex);
        float w1 = ex * w0;

        int p0 = atomicAdd(&counts[i0], 1);
        list_slot[i0 * CAP + p0] = token * 2 + 0;
        list_w  [i0 * CAP + p0] = w0;
        int p1 = atomicAdd(&counts[i1], 1);
        list_slot[i1 * CAP + p1] = token * 2 + 1;
        list_w  [i1 * CAP + p1] = w1;
    }
}

// ---------------------------------------------------------------- GEMM1
// hidden[slot, :] = gelu(X@W1[e] + b1[e]) * (X@W2[e] + b2[e])   (bf16 out)
// Block tile 64(M) x 128(N); 8 waves as 2(M) x 4(N); wave tile 32x32.
// Software pipelined: global loads for slab k+1 staged in regs during WMMA(k).
__global__ __launch_bounds__(256) void moe_gemm1_kernel(
    const float* __restrict__ x,  const float* __restrict__ W1,
    const float* __restrict__ b1, const float* __restrict__ W2,
    const float* __restrict__ b2, const int* __restrict__ counts,
    const int* __restrict__ list_slot, unsigned short* __restrict__ hidden) {
    const int e     = blockIdx.z;
    const int cnt   = counts[e];
    const int mbase = blockIdx.y * 64;
    if (mbase >= cnt) return;
    const int nbase = blockIdx.x * 128;

    __shared__ unsigned short ldsA [64  * RS];
    __shared__ unsigned short ldsB1[128 * RS];
    __shared__ unsigned short ldsB2[128 * RS];

    const int tid   = threadIdx.x;
    const int lane  = tid & 31;
    const int wv    = tid >> 5;
    const int waveM = wv & 1;    // 0..1
    const int waveN = wv >> 1;   // 0..3

    const float* W1e = W1 + (size_t)e * DM * DF;
    const float* W2e = W2 + (size_t)e * DM * DF;

    const v8f vz = {0.f,0.f,0.f,0.f,0.f,0.f,0.f,0.f};
    v8f acc1[2][2], acc2[2][2];
#pragma unroll
    for (int i = 0; i < 2; ++i)
#pragma unroll
        for (int j = 0; j < 2; ++j) { acc1[i][j] = vz; acc2[i][j] = vz; }

    // A gather assignment: 4 threads per row, 8 floats each
    const int arow  = tid >> 2;
    const int acoff = (tid & 3) * 8;
    const int agm   = mbase + arow;
    const int aslot = (agm < cnt) ? list_slot[e * CAP + agm] : 0;
    const float* asrc = x + (size_t)(aslot >> 1) * DM + acoff;

    // B assignment: 512 slots = 16 K-pairs x 32 N-quads, two slots per thread
    int kp[2], ng[2];
#pragma unroll
    for (int s = 0; s < 2; ++s) {
        int idx = tid + s * 256;
        kp[s] = idx >> 5;          // 0..15 (covers K rows 2kp, 2kp+1)
        ng[s] = (idx & 31) * 4;    // 0..124 (4 consecutive N)
    }

    // register staging for the pipelined K slab
    float4 sA0, sA1;
    float4 s1r0[2], s1r1[2], s2r0[2], s2r1[2];

    auto load_stage = [&](int k0) {
        sA0 = *(const float4*)(asrc + k0);
        sA1 = *(const float4*)(asrc + k0 + 4);
#pragma unroll
        for (int s = 0; s < 2; ++s) {
            const float* p1 = W1e + (size_t)(k0 + 2 * kp[s]) * DF + nbase + ng[s];
            const float* p2 = W2e + (size_t)(k0 + 2 * kp[s]) * DF + nbase + ng[s];
            s1r0[s] = *(const float4*)(p1);
            s1r1[s] = *(const float4*)(p1 + DF);
            s2r0[s] = *(const float4*)(p2);
            s2r1[s] = *(const float4*)(p2 + DF);
        }
    };

    auto store_stage = [&]() {
        uint4 pa;
        pa.x = pack_bf2(sA0.x, sA0.y);
        pa.y = pack_bf2(sA0.z, sA0.w);
        pa.z = pack_bf2(sA1.x, sA1.y);
        pa.w = pack_bf2(sA1.z, sA1.w);
        *(uint4*)(&ldsA[(size_t)arow * RS + acoff]) = pa;
#pragma unroll
        for (int s = 0; s < 2; ++s) {
            unsigned short* q1 = &ldsB1[(size_t)ng[s] * RS + 2 * kp[s]];
            unsigned short* q2 = &ldsB2[(size_t)ng[s] * RS + 2 * kp[s]];
            float a0[4] = {s1r0[s].x, s1r0[s].y, s1r0[s].z, s1r0[s].w};
            float a1[4] = {s1r1[s].x, s1r1[s].y, s1r1[s].z, s1r1[s].w};
            float c0[4] = {s2r0[s].x, s2r0[s].y, s2r0[s].z, s2r0[s].w};
            float c1[4] = {s2r1[s].x, s2r1[s].y, s2r1[s].z, s2r1[s].w};
#pragma unroll
            for (int u = 0; u < 4; ++u) {
                *(unsigned*)(q1 + (size_t)u * RS) = pack_bf2(a0[u], a1[u]); // K-pair packed
                *(unsigned*)(q2 + (size_t)u * RS) = pack_bf2(c0[u], c1[u]);
            }
        }
    };

    load_stage(0);
    for (int k0 = 0; k0 < DM; k0 += 32) {
        store_stage();
        __syncthreads();
        if (k0 + 32 < DM) {
            load_stage(k0 + 32);                 // overlaps with WMMA below
            if (k0 + 64 < DM) {
                __builtin_prefetch(W1e + (size_t)(k0 + 64 + 2 * kp[0]) * DF + nbase + ng[0], 0, 1);
                __builtin_prefetch(W2e + (size_t)(k0 + 64 + 2 * kp[0]) * DF + nbase + ng[0], 0, 1);
            }
        }

        v16bf af[2], bfa[2], bfb[2];
#pragma unroll
        for (int i = 0; i < 2; ++i) af[i] = ld_frag(ldsA, waveM * 32 + i * 16, lane);
#pragma unroll
        for (int j = 0; j < 2; ++j) {
            bfa[j] = ld_frag(ldsB1, waveN * 32 + j * 16, lane);
            bfb[j] = ld_frag(ldsB2, waveN * 32 + j * 16, lane);
        }
#pragma unroll
        for (int i = 0; i < 2; ++i)
#pragma unroll
            for (int j = 0; j < 2; ++j) {
                acc1[i][j] = __builtin_amdgcn_wmma_f32_16x16x32_bf16(
                    false, af[i], false, bfa[j], (short)0, acc1[i][j], false, false);
                acc2[i][j] = __builtin_amdgcn_wmma_f32_16x16x32_bf16(
                    false, af[i], false, bfb[j], (short)0, acc2[i][j], false, false);
            }
        __syncthreads();
    }

    // ---- epilogue: bias, exact GELU gate, bf16 store to hidden[slot]
    const int colb = nbase + waveN * 32 + (lane & 15);
    const int hh   = lane >> 4;
#pragma unroll
    for (int i = 0; i < 2; ++i) {
        int mloc = mbase + waveM * 32 + i * 16 + hh * 8;
#pragma unroll
        for (int r = 0; r < 8; ++r) {
            int gm = mloc + r;
            if (gm < cnt) {
                int slot = list_slot[e * CAP + gm];
#pragma unroll
                for (int j = 0; j < 2; ++j) {
                    int col  = colb + j * 16;
                    float h1 = acc1[i][j][r] + b1[e * DF + col];
                    float h2 = acc2[i][j][r] + b2[e * DF + col];
                    hidden[(size_t)slot * DF + col] = f2bf(gelu_exact(h1) * h2);
                }
            }
        }
    }
}

// ---------------------------------------------------------------- GEMM2
// out[tok] += w * (hidden[slot] @ W3[e] + b3[e])
// A (already bf16) is DMA'd by global_load_async_to_lds_b128, double-buffered
// so the copy of slab k+1 overlaps WMMA of slab k. B pipelined via registers.
__global__ __launch_bounds__(256) void moe_gemm2_kernel(
    const unsigned short* __restrict__ hidden, const float* __restrict__ W3,
    const float* __restrict__ b3, const int* __restrict__ counts,
    const int* __restrict__ list_slot, const float* __restrict__ list_w,
    float* __restrict__ out) {
    const int e     = blockIdx.z;
    const int cnt   = counts[e];
    const int mbase = blockIdx.y * 64;
    if (mbase >= cnt) return;
    const int nbase = blockIdx.x * 128;

    __shared__ unsigned short ldsA[2][64 * RS];
    __shared__ unsigned short ldsB[128 * RS];

    const int tid   = threadIdx.x;
    const int lane  = tid & 31;
    const int wv    = tid >> 5;
    const int waveM = wv & 1;
    const int waveN = wv >> 1;

    const float* W3e = W3 + (size_t)e * DF * DM;

    const v8f vz = {0.f,0.f,0.f,0.f,0.f,0.f,0.f,0.f};
    v8f acc[2][2];
#pragma unroll
    for (int i = 0; i < 2; ++i)
#pragma unroll
        for (int j = 0; j < 2; ++j) acc[i][j] = vz;

    const int arow  = tid >> 2;
    const int acoff = (tid & 3) * 8;
    const int agm   = mbase + arow;
    const int aslot = (agm < cnt) ? list_slot[e * CAP + agm] : 0;
    const unsigned short* asrc = hidden + (size_t)aslot * DF + acoff;

    const unsigned a_lds0 = (unsigned)(uintptr_t)&ldsA[0][(size_t)arow * RS + acoff];
    const unsigned a_lds1 = (unsigned)(uintptr_t)&ldsA[1][(size_t)arow * RS + acoff];

    int kp[2], ng[2];
#pragma unroll
    for (int s = 0; s < 2; ++s) {
        int idx = tid + s * 256;
        kp[s] = idx >> 5;
        ng[s] = (idx & 31) * 4;
    }

    float4 sbr0[2], sbr1[2];
    auto load_stageB = [&](int k0) {
#pragma unroll
        for (int s = 0; s < 2; ++s) {
            const float* p = W3e + (size_t)(k0 + 2 * kp[s]) * DM + nbase + ng[s];
            sbr0[s] = *(const float4*)(p);
            sbr1[s] = *(const float4*)(p + DM);
        }
    };
    auto store_stageB = [&]() {
#pragma unroll
        for (int s = 0; s < 2; ++s) {
            unsigned short* q = &ldsB[(size_t)ng[s] * RS + 2 * kp[s]];
            float a0[4] = {sbr0[s].x, sbr0[s].y, sbr0[s].z, sbr0[s].w};
            float a1[4] = {sbr1[s].x, sbr1[s].y, sbr1[s].z, sbr1[s].w};
#pragma unroll
            for (int u = 0; u < 4; ++u)
                *(unsigned*)(q + (size_t)u * RS) = pack_bf2(a0[u], a1[u]);
        }
    };

    // prologue: DMA A slab 0 into buffer 0; stage B slab 0 in regs
    async_copy_b128(a_lds0, asrc);
    load_stageB(0);

    for (int k0 = 0; k0 < DF; k0 += 32) {
        const int cur = (k0 >> 5) & 1;
        store_stageB();
        wait_async0();                 // A(cur) DMA complete before barrier
        __syncthreads();
        if (k0 + 32 < DF) {
            async_copy_b128(cur ? a_lds0 : a_lds1, asrc + k0 + 32); // overlaps WMMA
            load_stageB(k0 + 32);                                   // overlaps WMMA
            if (k0 + 64 < DF)
                __builtin_prefetch(W3e + (size_t)(k0 + 64 + 2 * kp[0]) * DM + nbase + ng[0], 0, 1);
        }

        v16bf af[2], bfr[2];
#pragma unroll
        for (int i = 0; i < 2; ++i) af[i]  = ld_frag(ldsA[cur], waveM * 32 + i * 16, lane);
#pragma unroll
        for (int j = 0; j < 2; ++j) bfr[j] = ld_frag(ldsB, waveN * 32 + j * 16, lane);
#pragma unroll
        for (int i = 0; i < 2; ++i)
#pragma unroll
            for (int j = 0; j < 2; ++j)
                acc[i][j] = __builtin_amdgcn_wmma_f32_16x16x32_bf16(
                    false, af[i], false, bfr[j], (short)0, acc[i][j], false, false);
        __syncthreads();
    }

    // ---- epilogue: bias, routing weight, atomic combine into out
    const int colb = nbase + waveN * 32 + (lane & 15);
    const int hh   = lane >> 4;
#pragma unroll
    for (int i = 0; i < 2; ++i) {
        int mloc = mbase + waveM * 32 + i * 16 + hh * 8;
#pragma unroll
        for (int r = 0; r < 8; ++r) {
            int gm = mloc + r;
            if (gm < cnt) {
                int slot = list_slot[e * CAP + gm];
                float w  = list_w[e * CAP + gm];
                int tok  = slot >> 1;
#pragma unroll
                for (int j = 0; j < 2; ++j) {
                    int col   = colb + j * 16;
                    float val = (acc[i][j][r] + b3[e * DM + col]) * w;
                    atomicAdd(&out[(size_t)tok * DM + col], val);
                }
            }
        }
    }
}

// ---------------------------------------------------------------- launch
extern "C" void kernel_launch(void* const* d_in, const int* in_sizes, int n_in,
                              void* d_out, int out_size, void* d_ws, size_t ws_size,
                              hipStream_t stream) {
    const float* x  = (const float*)d_in[0];
    const float* Wg = (const float*)d_in[1];
    const float* bg = (const float*)d_in[2];
    const float* W1 = (const float*)d_in[3];
    const float* b1 = (const float*)d_in[4];
    const float* W2 = (const float*)d_in[5];
    const float* b2 = (const float*)d_in[6];
    const float* W3 = (const float*)d_in[7];
    const float* b3 = (const float*)d_in[8];
    float* out = (float*)d_out;

    // workspace layout
    char* ws = (char*)d_ws;
    int*   counts    = (int*)ws;                                   // 32 B
    int*   list_slot = (int*)(ws + 256);                           // 64 KB
    float* list_w    = (float*)(ws + 256 + NE * CAP * 4);          // 64 KB
    unsigned short* hidden =
        (unsigned short*)(ws + 256 + 2 * NE * CAP * 4);            // 32 MB (4096 x 4096 bf16)

    moe_zero_kernel<<<512, 256, 0, stream>>>(out, NTOK * DM, counts);
    moe_router_kernel<<<NTOK / 8, 256, 0, stream>>>(x, Wg, bg, counts, list_slot, list_w);
    moe_gemm1_kernel<<<dim3(DF / 128, NTOK / 64, NE), 256, 0, stream>>>(
        x, W1, b1, W2, b2, counts, list_slot, hidden);
    moe_gemm2_kernel<<<dim3(DM / 128, NTOK / 64, NE), 256, 0, stream>>>(
        hidden, W3, b3, counts, list_slot, list_w, out);
}
